// TGAT_6030134083688
// MI455X (gfx1250) — matrix operand order repaired and checked by reading, not compiled, not run on hardware
//
#include <hip/hip_runtime.h>
#include <math.h>

// TGAT forward for MI455X (gfx1250, wave32).
// - Node projections (Q/K/V/skip, 32x32): v_wmma_f32_16x16x32_f16, one wave = 16 nodes.
// - Edge time-encoding matvec e = cos(...) @ We^T (the FLOP hot spot, 13 GFLOP over
//   2 passes) also on WMMA: one wave = 16 edges, cos() computed straight into the
//   A-operand layout, D-tile transposed through a per-wave LDS slice (stride 33).
// - Softmax normalization deferred to the output kernel (agg/denom), so the edge
//   phase is 2 passes. Node arrays (~64MB) are L2-resident (192MB L2).
// Workspace (floats): q[N*32] k[N*32] v[N*32] skip[N*32] agg[N*32]
//                     denom[N*2] umax[N*2] alpha[E*2]  (~92 MB total)

typedef __attribute__((ext_vector_type(16))) _Float16 v16h;
typedef __attribute__((ext_vector_type(8)))  float    v8f;

#define LSTRIDE 33  // LDS row stride (floats) for the 16x32 e-tile, conflict-breaking

__global__ void zero_ws_kernel(float* __restrict__ p, long long n) {
    long long i = (long long)blockIdx.x * blockDim.x + threadIdx.x;
    if (i < n) p[i] = 0.0f;
}

// ---------------- Node projections via WMMA ----------------

__device__ __forceinline__ void wmma_proj(v16h a,
                                          const float* __restrict__ W,
                                          const float* __restrict__ b,
                                          float* __restrict__ o,
                                          int m, int half_, int nb, int N,
                                          bool full) {
    // D[16 nodes x 16 cols] = A(h1 16x32) x B(W^T 32x16) + bias
    #pragma unroll
    for (int ct = 0; ct < 2; ++ct) {
        v16h bm;
        #pragma unroll
        for (int j = 0; j < 16; ++j)
            bm[j] = (_Float16)W[(ct * 16 + m) * 32 + (((j < 8) ? j : (j + 8)) + 8 * half_)];
        float bias = b[ct * 16 + m];
        v8f c;
        #pragma unroll
        for (int r = 0; r < 8; ++r) c[r] = bias;
        v8f d = __builtin_amdgcn_wmma_f32_16x16x32_f16(
            false, a, false, bm, (short)0, c, false, false);
        if (full) {
            #pragma unroll
            for (int r = 0; r < 8; ++r)
                o[(size_t)(nb + r + 8 * half_) * 32 + ct * 16 + m] = d[r];
        } else {
            #pragma unroll
            for (int r = 0; r < 8; ++r) {
                int nd = nb + r + 8 * half_;
                if (nd < N) o[(size_t)nd * 32 + ct * 16 + m] = d[r];
            }
        }
    }
}

__global__ __launch_bounds__(256)
void node_proj_kernel(const float* __restrict__ x,
                      const float* __restrict__ W_lin, const float* __restrict__ b_lin,
                      const float* __restrict__ Wq, const float* __restrict__ bq,
                      const float* __restrict__ Wk, const float* __restrict__ bk,
                      const float* __restrict__ Wv, const float* __restrict__ bv,
                      const float* __restrict__ Wsk, const float* __restrict__ bsk,
                      float* __restrict__ qn, float* __restrict__ kn,
                      float* __restrict__ vn, float* __restrict__ skn, int N) {
    const int lane  = threadIdx.x & 31;
    const int wave  = threadIdx.x >> 5;
    const int nb    = blockIdx.x * 128 + wave * 16;   // 8 waves x 16 nodes
    const int half_ = lane >> 4;
    const int m     = lane & 15;
    const int node  = nb + m;
    const bool full = (blockIdx.x * 128 + 128) <= N;  // block-uniform

    float xv[17];
    if (node < N) {
        #pragma unroll
        for (int i = 0; i < 17; ++i) xv[i] = x[(size_t)node * 17 + i];
    } else {
        #pragma unroll
        for (int i = 0; i < 17; ++i) xv[i] = 0.0f;
    }

    // h1 = relu(x @ W_lin^T + b_lin), produced directly into WMMA A layout
    v16h a;
    #pragma unroll
    for (int j = 0; j < 16; ++j) {
        int K = ((j < 8) ? j : (j + 8)) + 8 * half_;
        float acc = b_lin[K];
        const float* wr = W_lin + K * 17;
        #pragma unroll
        for (int i = 0; i < 17; ++i) acc = fmaf(xv[i], wr[i], acc);
        a[j] = (_Float16)fmaxf(acc, 0.0f);
    }

    wmma_proj(a, Wq,  bq,  qn,  m, half_, nb, N, full);
    wmma_proj(a, Wk,  bk,  kn,  m, half_, nb, N, full);
    wmma_proj(a, Wv,  bv,  vn,  m, half_, nb, N, full);
    wmma_proj(a, Wsk, bsk, skn, m, half_, nb, N, full);
}

// order-preserving float<->uint mapping for atomic max
__device__ __forceinline__ unsigned fmap(float f) {
    unsigned u = __float_as_uint(f);
    return (u & 0x80000000u) ? ~u : (u | 0x80000000u);
}
__device__ __forceinline__ float funmap(unsigned m) {
    return (m & 0x80000000u) ? __uint_as_float(m & 0x7FFFFFFFu)
                             : __uint_as_float(~m);
}

// ------------- Per-wave e-tile (16 edges x 32 cols) via WMMA -> LDS -------------
// Returns this lane's edge context (edge = eb0 + (lane&15), head = lane>>4).
// After this call, lds[m*LSTRIDE + half_*16 + j] holds e[edge m][head half_, j].

__device__ __forceinline__ void edge_etile(int eb0, int lane, int E,
                                           const int* __restrict__ ei,
                                           const float* __restrict__ t,
                                           const float* __restrict__ ntime,
                                           const float* __restrict__ bf,
                                           const float* __restrict__ ph,
                                           const float* __restrict__ We,
                                           const float* __restrict__ be,
                                           float* __restrict__ lds,
                                           int& s, int& d, bool& valid) {
    const int half_ = lane >> 4;
    const int m     = lane & 15;
    const int edge  = eb0 + m;
    valid = edge < E;
    s = valid ? ei[edge]     : 0;
    d = valid ? ei[E + edge] : 0;
    float rel = valid ? (ntime[s] - t[edge]) : 0.0f;

    // A = cos(rel*bf + ph) in WMMA A layout (row = edge m, K per slot)
    v16h a;
    #pragma unroll
    for (int j = 0; j < 16; ++j) {
        int K = ((j < 8) ? j : (j + 8)) + 8 * half_;
        a[j] = (_Float16)__cosf(fmaf(rel, bf[K], ph[K]));
    }
    // e = enc @ We^T + be, two 16-column tiles
    #pragma unroll
    for (int ct = 0; ct < 2; ++ct) {
        v16h bm;
        #pragma unroll
        for (int j = 0; j < 16; ++j)
            bm[j] = (_Float16)We[(ct * 16 + m) * 32 + (((j < 8) ? j : (j + 8)) + 8 * half_)];
        float bias = be[ct * 16 + m];
        v8f c;
        #pragma unroll
        for (int r = 0; r < 8; ++r) c[r] = bias;
        v8f dacc = __builtin_amdgcn_wmma_f32_16x16x32_f16(
            false, a, false, bm, (short)0, c, false, false);
        // D: VGPR r, lane: row = r + 8*half_, col = ct*16 + m  -> transpose via LDS
        #pragma unroll
        for (int r = 0; r < 8; ++r)
            lds[(r + 8 * half_) * LSTRIDE + ct * 16 + m] = dacc[r];
    }
    // same-wave LDS ops are in-order: no barrier needed before reading back
}

// ---------------- Edge pass 1: logits + segment max ----------------

__global__ __launch_bounds__(256)
void alpha_max_kernel(const int* __restrict__ ei, const float* __restrict__ t,
                      const float* __restrict__ ntime,
                      const float* __restrict__ bf, const float* __restrict__ ph,
                      const float* __restrict__ We, const float* __restrict__ be,
                      const float* __restrict__ qn, const float* __restrict__ kn,
                      float* __restrict__ alpha, unsigned* __restrict__ umax, int E) {
    __shared__ float etile[8 * 16 * LSTRIDE];
    const int lane = threadIdx.x & 31;
    const int wave = threadIdx.x >> 5;
    const int eb0  = (blockIdx.x * 8 + wave) * 16;
    float* lds = etile + wave * 16 * LSTRIDE;

    int s, d; bool valid;
    edge_etile(eb0, lane, E, ei, t, ntime, bf, ph, We, be, lds, s, d, valid);

    const int half_ = lane >> 4;
    const int m     = lane & 15;
    const int edge  = eb0 + m;
    const float* qr = qn + (size_t)d * 32 + half_ * 16;
    const float* kr = kn + (size_t)s * 32 + half_ * 16;
    const float* er = lds + m * LSTRIDE + half_ * 16;
    float acc = 0.0f;
    #pragma unroll
    for (int j = 0; j < 16; ++j) acc = fmaf(qr[j], kr[j] + er[j], acc);
    acc *= 0.25f;  // 1/sqrt(C), C=16
    if (valid) {
        alpha[2 * edge + half_] = acc;
        atomicMax(&umax[d * 2 + half_], fmap(acc));
    }
}

// ----- Edge pass 2: exp, denom, UNNORMALIZED weighted scatter-add -----

__global__ __launch_bounds__(256)
void msg_agg_kernel(const int* __restrict__ ei, const float* __restrict__ t,
                    const float* __restrict__ ntime,
                    const float* __restrict__ bf, const float* __restrict__ ph,
                    const float* __restrict__ We, const float* __restrict__ be,
                    const float* __restrict__ vn, const float* __restrict__ alpha,
                    const unsigned* __restrict__ umax,
                    float* __restrict__ denom, float* __restrict__ agg, int E) {
    __shared__ float etile[8 * 16 * LSTRIDE];
    const int lane = threadIdx.x & 31;
    const int wave = threadIdx.x >> 5;
    const int eb0  = (blockIdx.x * 8 + wave) * 16;
    float* lds = etile + wave * 16 * LSTRIDE;

    int s, d; bool valid;
    edge_etile(eb0, lane, E, ei, t, ntime, bf, ph, We, be, lds, s, d, valid);

    const int half_ = lane >> 4;
    const int m     = lane & 15;
    const int edge  = eb0 + m;
    if (valid) {
        float mx = funmap(umax[d * 2 + half_]);
        float a  = __expf(alpha[2 * edge + half_] - mx);
        atomicAdd(&denom[d * 2 + half_], a);
        const float* vr = vn + (size_t)s * 32 + half_ * 16;
        const float* er = lds + m * LSTRIDE + half_ * 16;
        #pragma unroll
        for (int j = 0; j < 16; ++j)
            atomicAdd(&agg[(size_t)d * 32 + half_ * 16 + j], (vr[j] + er[j]) * a);
    }
}

// ---------- Output: normalize, skip, 32->2, log_softmax ----------

__global__ __launch_bounds__(256)
void out_kernel(const float* __restrict__ agg, const float* __restrict__ denom,
                const float* __restrict__ skn,
                const float* __restrict__ W_out, const float* __restrict__ b_out,
                float* __restrict__ out, int N) {
    int n = blockIdx.x * blockDim.x + threadIdx.x;
    if (n >= N) return;
    float dh[2];
    #pragma unroll
    for (int h = 0; h < 2; ++h) {
        float dv = denom[n * 2 + h];
        dh[h] = (dv > 0.0f) ? (1.0f / dv) : 0.0f;  // isolated nodes: agg==0 anyway
    }
    float l[2];
    #pragma unroll
    for (int j = 0; j < 2; ++j) {
        float acc = b_out[j];
        #pragma unroll
        for (int c = 0; c < 32; ++c) {
            float hv = agg[(size_t)n * 32 + c] * dh[c >> 4] + skn[(size_t)n * 32 + c];
            acc = fmaf(hv, W_out[j * 32 + c], acc);
        }
        l[j] = acc;
    }
    float mx  = fmaxf(l[0], l[1]);
    float lse = mx + __logf(__expf(l[0] - mx) + __expf(l[1] - mx));
    out[2 * n]     = l[0] - lse;
    out[2 * n + 1] = l[1] - lse;
}

// ---------------- Host launch ----------------

extern "C" void kernel_launch(void* const* d_in, const int* in_sizes, int n_in,
                              void* d_out, int out_size, void* d_ws, size_t ws_size,
                              hipStream_t stream) {
    (void)n_in; (void)out_size; (void)ws_size;
    const float* x     = (const float*)d_in[0];
    const int*   ei    = (const int*)  d_in[1];
    const float* t     = (const float*)d_in[2];
    const float* ntime = (const float*)d_in[3];
    const float* bf    = (const float*)d_in[4];
    const float* ph    = (const float*)d_in[5];
    const float* W_lin = (const float*)d_in[6];
    const float* b_lin = (const float*)d_in[7];
    const float* Wq    = (const float*)d_in[8];
    const float* bq    = (const float*)d_in[9];
    const float* Wk    = (const float*)d_in[10];
    const float* bk    = (const float*)d_in[11];
    const float* Wv    = (const float*)d_in[12];
    const float* bv    = (const float*)d_in[13];
    const float* We    = (const float*)d_in[14];
    const float* be    = (const float*)d_in[15];
    const float* Wsk   = (const float*)d_in[16];
    const float* bsk   = (const float*)d_in[17];
    const float* W_out = (const float*)d_in[18];
    const float* b_out = (const float*)d_in[19];
    float* out = (float*)d_out;

    const int N = in_sizes[3];   // node_time length
    const int E = in_sizes[2];   // t length
    const size_t N32 = (size_t)N * 32;

    float* qn    = (float*)d_ws;
    float* kn    = qn  + N32;
    float* vn    = kn  + N32;
    float* skn   = vn  + N32;
    float* agg   = skn + N32;
    float* denom = agg + N32;
    unsigned* umax = (unsigned*)(denom + (size_t)2 * N);
    float* alpha = (float*)(umax + (size_t)2 * N);

    // zero agg | denom | umax (contiguous)
    long long zc = (long long)N32 + 4LL * N;
    zero_ws_kernel<<<(int)((zc + 255) / 256), 256, 0, stream>>>(agg, zc);

    node_proj_kernel<<<(N + 127) / 128, 256, 0, stream>>>(
        x, W_lin, b_lin, Wq, bq, Wk, bk, Wv, bv, Wsk, bsk, qn, kn, vn, skn, N);

    int eb = (E + 127) / 128;  // 128 edges per block (8 waves x 16 edges)
    alpha_max_kernel<<<eb, 256, 0, stream>>>(ei, t, ntime, bf, ph, We, be,
                                             qn, kn, alpha, umax, E);
    msg_agg_kernel<<<eb, 256, 0, stream>>>(ei, t, ntime, bf, ph, We, be,
                                           vn, alpha, umax, denom, agg, E);
    out_kernel<<<(N + 255) / 256, 256, 0, stream>>>(agg, denom, skn, W_out, b_out,
                                                    out, N);
}
